// ShiftHeadAttention_54735063220788
// MI455X (gfx1250) — compile-verified, matrix operand order checked by reading, not run on hardware
//
#include <hip/hip_runtime.h>
#include <hip/hip_bf16.h>

// ---------------------------------------------------------------------------
// ShiftHeadAttention for MI455X (gfx1250, wave32, WMMA).
// B=16, QH=QW=64, KH=KW=64, H=8, D=256, DK=32, K=4, S=1.
// GEMMs: V_WMMA_F32_16X16X32_BF16 with bf16 hi/lo (bf16x3) splitting for
// fp32-class accuracy. Weights pre-split+permuted to fragment order so the
// B operand streams global->LDS with async copies; activations are split
// once at LDS staging. Sampling is an L2-resident gather (kf = 64MB << 192MB).
// ---------------------------------------------------------------------------

typedef __bf16 bf16_t;
typedef __bf16 v16bf __attribute__((ext_vector_type(16)));
typedef __bf16 v8bfv __attribute__((ext_vector_type(8)));
typedef __bf16 v4bf  __attribute__((ext_vector_type(4)));
typedef float  v8f   __attribute__((ext_vector_type(8)));

#define NB 16
#define NQH 64
#define NQW 64
#define NKH 64
#define NKW 64
#define NHEAD 8
#define NDIM 256
#define NDK 32
#define NPTS 4
#define NQ (NB * NQH * NQW)            /* 65536 query positions  */
#define NKPIX (NB * NKH * NKW)         /* 65536 key positions    */

#if defined(__has_builtin)
#  if __has_builtin(__builtin_amdgcn_global_load_async_to_lds_b128) && \
      __has_builtin(__builtin_amdgcn_s_wait_asynccnt)
#    define USE_ASYNC_LDS 1
#  endif
#endif
#ifndef USE_ASYNC_LDS
#  define USE_ASYNC_LDS 0
#endif

// Types matching the async-LDS builtin signature:
//   void __builtin_amdgcn_global_load_async_to_lds_b128(v4i AS1*, v4i AS3*, Ii, Ii)
typedef int v4i_g __attribute__((vector_size(16)));
typedef __attribute__((address_space(1))) v4i_g as1_v4i;
typedef __attribute__((address_space(3))) v4i_g as3_v4i;

// Swap bits 3 and 4 of a 5-bit k index: maps K order [0..7,8..15,16..23,24..31]
// to fragment storage order [0..7,16..23,8..15,24..31] (ISA 16-bit 16x32 layout:
// lanes 0-15 hold K {0..7,16..23}, lanes 16-31 hold K {8..15,24..31}).
__device__ __forceinline__ int permk(int k) {
  return (k & ~24) | ((k & 8) << 1) | ((k & 16) >> 1);
}

// 16-byte global -> LDS copy; async (ASYNCcnt-tracked) when available.
__device__ __forceinline__ void copy16_to_lds(bf16_t* dst, const bf16_t* src) {
#if USE_ASYNC_LDS
  __builtin_amdgcn_global_load_async_to_lds_b128((as1_v4i*)src, (as3_v4i*)dst, 0, 0);
#else
  *(float4*)dst = *(const float4*)src;
#endif
}

__device__ __forceinline__ void wait_async_then_barrier() {
#if USE_ASYNC_LDS
  __builtin_amdgcn_s_wait_asynccnt(0);
#endif
  __syncthreads();
}

// Load a 16-element bf16 fragment slice from LDS (16B-aligned halves).
__device__ __forceinline__ v16bf load_frag(const bf16_t* p) {
  v8bfv a = *(const v8bfv*)(p);
  v8bfv b = *(const v8bfv*)(p + 8);
  return __builtin_shufflevector(a, b, 0, 1, 2, 3, 4, 5, 6, 7,
                                 8, 9, 10, 11, 12, 13, 14, 15);
}

// One-shot weight prep: W[256][N] fp32 -> whi/wlo[N][256] bf16, k permuted to
// fragment order so GEMM B staging is a straight 16B copy.
__global__ __launch_bounds__(256)
void wsplit_kernel(const float* __restrict__ W, bf16_t* __restrict__ whi,
                   bf16_t* __restrict__ wlo, int N) {
  int t = blockIdx.x * 256 + threadIdx.x;
  if (t >= N * 256) return;
  int n = t >> 8;
  int k = t & 255;
  float x = W[(size_t)k * N + n];
  bf16_t h = (bf16_t)x;
  bf16_t l = (bf16_t)(x - (float)h);
  int kp = (k & ~31) | permk(k & 31);
  whi[(size_t)n * 256 + kp] = h;
  wlo[(size_t)n * 256 + kp] = l;
}

// C[M,N] = A[M,256] * W[256,N] + bias[N]; A fp32, W pre-split bf16 hi/lo.
// BM=128, 256 threads (8 wave32s) arranged WM_ x WN_; each wave owns TM x TN
// 16x16 tiles. bf16x3 accumulation: hi*hi + hi*lo + lo*hi in f32.
template <int BN, int WM_, int WN_, int TM, int TN>
__global__ __launch_bounds__(256)
void gemm_bias_bf16x3(const float* __restrict__ A, const bf16_t* __restrict__ whi,
                      const bf16_t* __restrict__ wlo, const float* __restrict__ bias,
                      float* __restrict__ C, int M, int N) {
  constexpr int BM = 128, KB = 32, KTOT = 256, LDT = KB + 8;  // stride 40 bf16 = 80B
  __shared__ bf16_t sAhi[BM * LDT];
  __shared__ bf16_t sAlo[BM * LDT];
  __shared__ bf16_t sBhi[BN * LDT];
  __shared__ bf16_t sBlo[BN * LDT];

  const int tid   = threadIdx.x;
  const int lane  = tid & 31;
  const int wave  = tid >> 5;
  const int waveM = wave % WM_;
  const int waveN = wave / WM_;
  const int wmb   = waveM * (TM * 16);
  const int wnb   = waveN * (TN * 16);
  const int row0  = blockIdx.x * BM;
  const int col0  = blockIdx.y * BN;
  const int fr    = lane & 15;               // fragment row (A) / col (B)
  const int halfo = (lane & 16) ? 16 : 0;    // K-half select (permuted order)

  v8f acc[TM][TN];
#pragma unroll
  for (int i = 0; i < TM; ++i)
#pragma unroll
    for (int j = 0; j < TN; ++j)
      acc[i][j] = (v8f){0.f, 0.f, 0.f, 0.f, 0.f, 0.f, 0.f, 0.f};

  for (int kb = 0; kb < KTOT; kb += KB) {
    // ---- stage A tile: load f32, split to bf16 hi/lo, store permuted ----
#pragma unroll
    for (int i = 0; i < (BM * KB / 4) / 256; ++i) {
      int idx = tid + i * 256;
      int r = idx >> 3;
      int c4 = (idx & 7) << 2;                 // k offset within 32, mult of 4
      float4 v = *(const float4*)(A + (size_t)(row0 + r) * KTOT + kb + c4);
      int kp = permk(c4);                      // whole float4 stays in one 8-group
      float f[4] = {v.x, v.y, v.z, v.w};
      v4bf h4, l4;
#pragma unroll
      for (int j = 0; j < 4; ++j) {
        bf16_t h = (bf16_t)f[j];
        h4[j] = h;
        l4[j] = (bf16_t)(f[j] - (float)h);
      }
      *(v4bf*)(&sAhi[r * LDT + kp]) = h4;
      *(v4bf*)(&sAlo[r * LDT + kp]) = l4;
      if (kb + KB < KTOT)  // hint next k-slice of A (global_prefetch_b8)
        __builtin_prefetch((const void*)(A + (size_t)(row0 + r) * KTOT + kb + KB + c4), 0, 1);
    }
    // ---- stage B tile: straight 16B copies (async -> LDS when available) ----
#pragma unroll
    for (int i = 0; i < (BN * 8) / 256; ++i) { // BN*4 chunks/plane, 2 planes
      int idx = tid + i * 256;
      const bf16_t* plane = (idx < BN * 4) ? whi : wlo;
      bf16_t* splane      = (idx < BN * 4) ? sBhi : sBlo;
      int id2 = idx & (BN * 4 - 1);
      int n = id2 >> 2;
      int c = (id2 & 3) * 8;                   // 8 bf16 = 16B
      copy16_to_lds(&splane[n * LDT + c],
                    plane + (size_t)(col0 + n) * 256 + kb + c);
    }
    wait_async_then_barrier();

    v16bf ahi[TM], alo[TM];
#pragma unroll
    for (int tm = 0; tm < TM; ++tm) {
      int r = wmb + tm * 16 + fr;
      ahi[tm] = load_frag(&sAhi[r * LDT + halfo]);
      alo[tm] = load_frag(&sAlo[r * LDT + halfo]);
    }
#pragma unroll
    for (int tn = 0; tn < TN; ++tn) {
      int n = wnb + tn * 16 + fr;
      v16bf bhi = load_frag(&sBhi[n * LDT + halfo]);
      v16bf blo = load_frag(&sBlo[n * LDT + halfo]);
#pragma unroll
      for (int tm = 0; tm < TM; ++tm) {
        acc[tm][tn] = __builtin_amdgcn_wmma_f32_16x16x32_bf16(
            false, ahi[tm], false, bhi, (short)0, acc[tm][tn], false, false);
        acc[tm][tn] = __builtin_amdgcn_wmma_f32_16x16x32_bf16(
            false, ahi[tm], false, blo, (short)0, acc[tm][tn], false, false);
        acc[tm][tn] = __builtin_amdgcn_wmma_f32_16x16x32_bf16(
            false, alo[tm], false, bhi, (short)0, acc[tm][tn], false, false);
      }
    }
    __syncthreads();
  }

  // ---- epilogue: bias + store (C/D: elem v = row v (+8 for upper half-wave))
#pragma unroll
  for (int tn = 0; tn < TN; ++tn) {
    int col = col0 + wnb + tn * 16 + fr;
    float bv = bias[col];
#pragma unroll
    for (int tm = 0; tm < TM; ++tm) {
      int mr = row0 + wmb + tm * 16 + ((lane & 16) ? 8 : 0);
#pragma unroll
      for (int v = 0; v < 8; ++v)
        C[(size_t)(mr + v) * N + col] = acc[tm][tn][v] + bv;
    }
  }
}

// Softmax over contiguous groups of 4 (per query, per head), in place.
__global__ __launch_bounds__(256)
void softmax4_kernel(float* __restrict__ Ab, int total) {
  int i = blockIdx.x * blockDim.x + threadIdx.x;
  if (i >= total) return;
  float* p = Ab + (size_t)i * 4;
  float a0 = p[0], a1 = p[1], a2 = p[2], a3 = p[3];
  float m = fmaxf(fmaxf(a0, a1), fmaxf(a2, a3));
  float e0 = __expf(a0 - m), e1 = __expf(a1 - m);
  float e2 = __expf(a2 - m), e3 = __expf(a3 - m);
  float inv = 1.0f / (e0 + e1 + e2 + e3);
  p[0] = e0 * inv; p[1] = e1 * inv; p[2] = e2 * inv; p[3] = e3 * inv;
}

// One wave32 per (query, head); lane = DK channel. Bilinear zero-padded
// sampling of kf ([B, KH, KW, H*DK] straight from the GEMM), weighted by
// softmaxed attention.
__global__ __launch_bounds__(256)
void sample_kernel(const float* __restrict__ kf, const float* __restrict__ offb,
                   const float* __restrict__ Ab, const float* __restrict__ ref,
                   float* __restrict__ feat) {
  const int q = blockIdx.x;
  const int h = threadIdx.x >> 5;
  const int d = threadIdx.x & 31;
  const int b = q >> 12;
  const int qy = (q >> 6) & 63;
  const int qx = q & 63;
  // Faithful reference quirk: rr tiled over H then indexed by n=b*H+h -> n % B
  const int rb = (b * NHEAD + h) & (NB - 1);
  const int rq = ((rb * NQH + qy) * NQW + qx);
  const float rx = ref[(size_t)rq * 2 + 0] * (float)(NKW - 1);
  const float ry = ref[(size_t)rq * 2 + 1] * (float)(NKH - 1);
  const float* offp = offb + (size_t)q * (2 * NHEAD * NPTS) + h * (2 * NPTS);
  const float* Ap   = Ab   + (size_t)q * (NHEAD * NPTS)     + h * NPTS;
  const float* img  = kf + (size_t)b * NKH * NKW * NDIM + h * NDK + d;

  float accv = 0.0f;
#pragma unroll
  for (int k = 0; k < NPTS; ++k) {
    float ptx = rx + offp[k * 2 + 0];
    float pty = ry + offp[k * 2 + 1];
    float vgx = ptx * (2.0f / (float)(NKW - 1)) - 1.0f;
    float vgy = pty * (2.0f / (float)(NKH - 1)) - 1.0f;
    float px = ((vgx + 1.0f) * (float)NKW - 1.0f) * 0.5f;
    float py = ((vgy + 1.0f) * (float)NKH - 1.0f) * 0.5f;
    float x0f = floorf(px), y0f = floorf(py);
    int x0 = (int)x0f, y0 = (int)y0f;
    float wx1 = px - x0f, wy1 = py - y0f;
    float wx[2] = {1.0f - wx1, wx1};
    float wy[2] = {1.0f - wy1, wy1};
    int xs[2] = {x0, x0 + 1};
    int ys[2] = {y0, y0 + 1};
    float s = 0.0f;
#pragma unroll
    for (int cy = 0; cy < 2; ++cy)
#pragma unroll
      for (int cx = 0; cx < 2; ++cx) {
        int xi = xs[cx], yi = ys[cy];
        if (xi >= 0 && xi < NKW && yi >= 0 && yi < NKH)
          s += wx[cx] * wy[cy] * img[(size_t)(yi * NKW + xi) * NDIM];
      }
    accv += Ap[k] * s;
  }
  feat[(size_t)q * NDIM + h * NDK + d] = accv;
}

extern "C" void kernel_launch(void* const* d_in, const int* in_sizes, int n_in,
                              void* d_out, int out_size, void* d_ws, size_t ws_size,
                              hipStream_t stream) {
  // 0 query (UNUSED: q_proj is dead in the reference), 1 keys, 2 ref_point,
  // 3 src_query, 4 Wq (unused), 5 bq (unused), 6 Wb, 7 bb, 8 Wk, 9 bk,
  // 10 Woff, 11 boff, 12 WA, 13 bA, 14 Wm, 15 bm
  const float* keys = (const float*)d_in[1];
  const float* ref  = (const float*)d_in[2];
  const float* srcq = (const float*)d_in[3];
  const float* Wb   = (const float*)d_in[6];
  const float* bb   = (const float*)d_in[7];
  const float* Wk   = (const float*)d_in[8];
  const float* bk   = (const float*)d_in[9];
  const float* Woff = (const float*)d_in[10];
  const float* boff = (const float*)d_in[11];
  const float* WA   = (const float*)d_in[12];
  const float* bA   = (const float*)d_in[13];
  const float* Wm   = (const float*)d_in[14];
  const float* bm   = (const float*)d_in[15];
  float* out = (float*)d_out;

  // Workspace carve-up. feat reuses sq (dead after off/A GEMMs).
  float* ws   = (float*)d_ws;
  float* sq   = ws;                                   // [NQ, 256]
  float* kfb  = sq   + (size_t)NQ * NDIM;             // [NKPIX, 256]
  float* offb = kfb  + (size_t)NKPIX * NDIM;          // [NQ, 64]
  float* Ab   = offb + (size_t)NQ * 2 * NHEAD * NPTS; // [NQ, 32]
  float* feat = sq;                                   // reuse
  bf16_t* wsp = (bf16_t*)(Ab + (size_t)NQ * NHEAD * NPTS);
  bf16_t* wb_s   = wsp;                               // [2][256][256]
  bf16_t* wk_s   = wb_s   + 2 * 256 * 256;
  bf16_t* wm_s   = wk_s   + 2 * 256 * 256;
  bf16_t* woff_s = wm_s   + 2 * 256 * 256;            // [2][64][256]
  bf16_t* wa_s   = woff_s + 2 * 64 * 256;             // [2][32][256]

  const dim3 blk(256);

  // 0) pre-split all weight matrices into permuted bf16 hi/lo planes
  wsplit_kernel<<<dim3((256 * 256 + 255) / 256), blk, 0, stream>>>(Wb, wb_s, wb_s + 256 * 256, 256);
  wsplit_kernel<<<dim3((256 * 256 + 255) / 256), blk, 0, stream>>>(Wk, wk_s, wk_s + 256 * 256, 256);
  wsplit_kernel<<<dim3((256 * 256 + 255) / 256), blk, 0, stream>>>(Wm, wm_s, wm_s + 256 * 256, 256);
  wsplit_kernel<<<dim3((64 * 256 + 255) / 256), blk, 0, stream>>>(Woff, woff_s, woff_s + 64 * 256, 64);
  wsplit_kernel<<<dim3((32 * 256 + 255) / 256), blk, 0, stream>>>(WA, wa_s, wa_s + 32 * 256, 32);

  // 1) sq = src_query @ Wb + bb            [65536,256]x[256,256], A read once
  gemm_bias_bf16x3<256, 2, 4, 4, 4><<<dim3(NQ / 128, 1), blk, 0, stream>>>(
      srcq, wb_s, wb_s + 256 * 256, bb, sq, NQ, NDIM);
  // 2) off = sq @ Woff + boff              [65536,256]x[256,64]
  gemm_bias_bf16x3<64, 8, 1, 1, 4><<<dim3(NQ / 128, 1), blk, 0, stream>>>(
      sq, woff_s, woff_s + 64 * 256, boff, offb, NQ, 2 * NHEAD * NPTS);
  // 3) A = sq @ WA + bA                    [65536,256]x[256,32]
  gemm_bias_bf16x3<32, 8, 1, 1, 2><<<dim3(NQ / 128, 1), blk, 0, stream>>>(
      sq, wa_s, wa_s + 32 * 256, bA, Ab, NQ, NHEAD * NPTS);
  // 4) softmax over each group of 4
  {
    int total = NQ * NHEAD;
    softmax4_kernel<<<dim3((total + 255) / 256), blk, 0, stream>>>(Ab, total);
  }
  // 5) kf = keys @ Wk + bk                 [65536,256]x[256,256]
  gemm_bias_bf16x3<256, 2, 4, 4, 4><<<dim3(NKPIX / 128, 1), blk, 0, stream>>>(
      keys, wk_s, wk_s + 256 * 256, bk, kfb, NKPIX, NDIM);
  // 6) bilinear sampling + attention-weighted sum -> feat
  sample_kernel<<<dim3(NQ), blk, 0, stream>>>(kfb, offb, Ab, ref, feat);
  // 7) out = feat @ Wm + bm                [65536,256]x[256,256]
  gemm_bias_bf16x3<256, 2, 4, 4, 4><<<dim3(NQ / 128, 1), blk, 0, stream>>>(
      feat, wm_s, wm_s + 256 * 256, bm, out, NQ, NDIM);
}
